// WaveInterferenceLayer_65171833749648
// MI455X (gfx1250) — matrix operand order repaired and checked by reading, not compiled, not run on hardware
//
#include <hip/hip_runtime.h>
#include <hip/hip_bf16.h>
#include <cstdint>

// ---------------- problem constants (match reference) ----------------
static constexpr int   Bb   = 8;
static constexpr int   Ss   = 4096;
static constexpr int   Hh   = 1024;
static constexpr int   Hhalf= 512;
static constexpr int   Aa   = 512;      // NUM_ANT
static constexpr int   TOK  = Bb * Ss;  // 32768 tokens
static constexpr float SPEED_INV = 1.0f / 100.0f;
static constexpr float DECAY_C   = 0.1f;
static constexpr float MAX_FREQ_C= 10.0f;
static constexpr float PI_F      = 3.14159265358979f;
static constexpr float EPS_LN    = 1e-5f;

// ---------------- GEMM tiling ----------------
static constexpr int MT       = 64;            // tokens per block
static constexpr int LDS_ROWB = 2064;          // bytes per padded LDS row (2048 + 16 pad)
static constexpr int NCHUNK   = 96;            // 3 MLPs * (512/16) n-chunks
static constexpr int KSLABS   = 32;            // 1024 / 32

typedef __attribute__((ext_vector_type(16))) __bf16 bf16x16;
typedef __attribute__((ext_vector_type(8)))  float  floatx8;
typedef __attribute__((ext_vector_type(4)))  unsigned int v4u;
typedef __attribute__((ext_vector_type(8)))  int          v8i;
typedef __attribute__((ext_vector_type(4)))  int          v4i;

union FragAB { bf16x16 v; uint4 q[2]; };
union Pack16 { __bf16 h[16]; uint4 q[2]; };
union Pack8  { __bf16 h[8];  uint4 q;    };

// =====================================================================
// Kernel 0a: hidden_states fp32 -> contiguous bf16 copy (TDM source).
// =====================================================================
__global__ void cvt_hs_kernel(const float* __restrict__ hs, __bf16* __restrict__ hsbf)
{
    size_t i = ((size_t)blockIdx.x * 256 + threadIdx.x) * 8;
    const float4* s = (const float4*)(hs + i);
    float4 a = s[0], b = s[1];
    Pack8 p;
    p.h[0] = (__bf16)a.x; p.h[1] = (__bf16)a.y; p.h[2] = (__bf16)a.z; p.h[3] = (__bf16)a.w;
    p.h[4] = (__bf16)b.x; p.h[5] = (__bf16)b.y; p.h[6] = (__bf16)b.z; p.h[7] = (__bf16)b.w;
    *(uint4*)(hsbf + i) = p.q;
}

// =====================================================================
// Kernel 0b: pack the three (1024 x 512) fp32 W1 matrices into bf16
// WMMA B-fragment order: fragment id f = (chunk*32 + kslab)*32 + lane.
// =====================================================================
__global__ void pack_w1_kernel(const float* __restrict__ aw1,
                               const float* __restrict__ fw1,
                               const float* __restrict__ pw1,
                               __bf16* __restrict__ pw)
{
    int fid  = blockIdx.x * 256 + threadIdx.x;   // 0 .. 98303
    int lane = fid & 31;
    int k0   = (fid >> 5) & 31;                  // k-slab
    int c    = fid >> 10;                        // chunk 0..95
    int mlp  = c >> 5;
    int n    = ((c & 31) << 4) + (lane & 15);
    int kbase= k0 * 32 + (lane >> 4) * 8;
    const float* w1 = (mlp == 0) ? aw1 : ((mlp == 1) ? fw1 : pw1);

    Pack16 p;
#pragma unroll
    for (int j = 0; j < 16; ++j)
        p.h[j] = (__bf16)w1[(size_t)(kbase + j + (j >= 8 ? 8 : 0)) * Hhalf + n];

    uint4* dst = (uint4*)(pw + (size_t)fid * 16);
    dst[0] = p.q[0];
    dst[1] = p.q[1];
}

// =====================================================================
// Kernel 1: fused 3-MLP GEMM. Block = 256 threads (8 waves), 64 tokens.
// A tile staged by the Tensor Data Mover into padded LDS rows.
// =====================================================================
__global__ void mlp_gemm_kernel(const __bf16* __restrict__ hsbf,
                                const __bf16* __restrict__ pw,
                                const float* __restrict__ ab1, const float* __restrict__ fb1, const float* __restrict__ pb1,
                                const float* __restrict__ aw2, const float* __restrict__ fw2, const float* __restrict__ pw2,
                                const float* __restrict__ ab2, const float* __restrict__ fb2, const float* __restrict__ pb2,
                                float* __restrict__ ampo, float* __restrict__ freqo, float* __restrict__ pho)
{
    extern __shared__ char smem[];
    char*  Alds = smem;                           // MT * LDS_ROWB bytes, at LDS offset 0
    float* scal = (float*)(smem + MT * LDS_ROWB); // 3 * 64 floats

    const int tid = threadIdx.x;
    const int blk = blockIdx.x;
    const int lane = tid & 31;
    const int wave = tid >> 5;

    if (tid < 192) scal[tid] = 0.0f;

    const uint64_t ga = (uint64_t)(uintptr_t)(hsbf + (size_t)blk * MT * Hh);

#if __has_builtin(__builtin_amdgcn_tensor_load_to_lds)
    // ---- TDM: one iterated descriptor copies the whole 64x2048B tile ----
    // tile = 1 row of 256 x 8B; 64 iterations; LDS advances 258*8 = 2064B/row
    if (wave == 0) {
        v4u g0; v8i g1; v4i g2; v4i g3;
        g0[0] = 1u;                                   // count=1, user mode
        g0[1] = 0u;                                   // lds_addr = 0 (tile base)
        g0[2] = (unsigned int)(ga & 0xFFFFFFFFull);   // global_addr[31:0]
        g0[3] = (unsigned int)((ga >> 32) & 0x01FFFFFFull) | (2u << 30); // addr[56:32] | type=2

        g1[0] = (int)((3u << 16) | (1u << 19));       // data_size=8B, iterate_enable
        g1[1] = (int)(256u << 16);                    // tensor_dim0 = 256 (low 16 bits)
        g1[2] = (int)(1u << 16);                      // tensor_dim0 hi=0 | tensor_dim1 = 1
        g1[3] = (int)(256u << 16);                    // tensor_dim1 hi=0 | tile_dim0 = 256
        g1[4] = 1;                                    // tile_dim1 = 1, tile_dim2 = 0
        g1[5] = 256;                                  // tensor_dim0_stride = 256
        g1[6] = 0;
        g1[7] = 0;

        g2[0] = 1;                                    // tensor_dim2 (unused)
        g2[1] = 258;                                  // lds_addr_increment  = 2064B / 8
        g2[2] = 256;                                  // global_addr_increment = 2048B / 8
        g2[3] = (int)(63u << 16);                     // iterate_count = 63 (64 rows)

        g3[0] = 0; g3[1] = 0; g3[2] = 0; g3[3] = 0;

#if defined(__clang_major__) && (__clang_major__ >= 23)
        v8i gz; gz[0]=0;gz[1]=0;gz[2]=0;gz[3]=0;gz[4]=0;gz[5]=0;gz[6]=0;gz[7]=0;
        __builtin_amdgcn_tensor_load_to_lds(g0, g1, g2, g3, gz, 0);
#else
        __builtin_amdgcn_tensor_load_to_lds(g0, g1, g2, g3, 0);
#endif
#if __has_builtin(__builtin_amdgcn_s_wait_tensorcnt)
        __builtin_amdgcn_s_wait_tensorcnt(0);
#else
        asm volatile("s_wait_tensorcnt 0x0" ::: "memory");
#endif
    }
#else
    // ---- fallback: cooperative bf16 copy into padded LDS rows ----
    {
        const uint4* src = (const uint4*)(uintptr_t)ga;
#pragma unroll
        for (int i = 0; i < 32; ++i) {
            int idx = tid + i * 256;      // 8192 x 16B chunks
            int r   = idx >> 7;
            int c16 = idx & 127;
            uint4 v = src[idx];
            *(uint4*)(Alds + (size_t)r * LDS_ROWB + c16 * 16) = v;
        }
    }
#endif
    __syncthreads();

    const int half = lane >> 4;
    const int ncol = lane & 15;
    const uint4* bq = (const uint4*)pw;

    // per-mt LDS fragment base pointers (q[0] at +0, q[1] at +2, k-slab step = +4)
    const uint4* arow[4];
#pragma unroll
    for (int mt = 0; mt < 4; ++mt)
        arow[mt] = (const uint4*)(Alds + (size_t)(mt * 16 + ncol) * LDS_ROWB + half * 16);

    for (int c = wave; c < NCHUNK; c += 8) {
        const int mlp   = c >> 5;
        const int nbase = (c & 31) << 4;
        const int n     = nbase + ncol;

        floatx8 acc[4];
#pragma unroll
        for (int mt = 0; mt < 4; ++mt)
#pragma unroll
            for (int v = 0; v < 8; ++v) acc[mt][v] = 0.0f;

        const size_t fbase = ((size_t)c * KSLABS * 32 + lane) * 2; // uint4 index at k0=0

        // hint-prefetch the next chunk's first B fragments
        if (c + 8 < NCHUNK)
            __builtin_prefetch(&bq[((size_t)(c + 8) * KSLABS * 32 + lane) * 2], 0, 0);

        // ---- double-buffered K loop: loads for slab k+1 overlap WMMAs of slab k ----
        FragAB afr[2][4];
        FragAB bfr[2];

        bfr[0].q[0] = bq[fbase];
        bfr[0].q[1] = bq[fbase + 1];
#pragma unroll
        for (int mt = 0; mt < 4; ++mt) {
            afr[0][mt].q[0] = arow[mt][0];
            afr[0][mt].q[1] = arow[mt][2];
        }

#pragma unroll 4
        for (int k0 = 0; k0 < KSLABS; ++k0) {
            const int cur = k0 & 1;
            const int nxt = cur ^ 1;
            if (k0 + 1 < KSLABS) {
                size_t bi = fbase + (size_t)(k0 + 1) * 64;
                bfr[nxt].q[0] = bq[bi];
                bfr[nxt].q[1] = bq[bi + 1];
#pragma unroll
                for (int mt = 0; mt < 4; ++mt) {
                    afr[nxt][mt].q[0] = arow[mt][(k0 + 1) * 4];
                    afr[nxt][mt].q[1] = arow[mt][(k0 + 1) * 4 + 2];
                }
            }
#pragma unroll
            for (int mt = 0; mt < 4; ++mt) {
                acc[mt] = __builtin_amdgcn_wmma_f32_16x16x32_bf16(
                    false, afr[cur][mt].v, false, bfr[cur].v, (short)0, acc[mt], false, false);
            }
        }

        // ---- epilogue: bias + ReLU + dot(w2) folded into per-token scalars ----
        const float* b1p = (mlp == 0) ? ab1 : ((mlp == 1) ? fb1 : pb1);
        const float* w2p = (mlp == 0) ? aw2 : ((mlp == 1) ? fw2 : pw2);
        const float b1v = b1p[n];
        const float w2v = w2p[n];
#pragma unroll
        for (int mt = 0; mt < 4; ++mt) {
#pragma unroll
            for (int v = 0; v < 8; ++v) {
                float h = acc[mt][v] + b1v;
                h = fmaxf(h, 0.0f) * w2v;
                h += __shfl_xor(h, 1, 32);
                h += __shfl_xor(h, 2, 32);
                h += __shfl_xor(h, 4, 32);
                h += __shfl_xor(h, 8, 32);
                if (ncol == 0)
                    atomicAdd(&scal[mlp * 64 + mt * 16 + half * 8 + v], h);
            }
        }
    }
    __syncthreads();

    if (tid < 64) {
        int t = blk * MT + tid;
        float sa = scal[0 * 64 + tid] + ab2[0];
        float sf = scal[1 * 64 + tid] + fb2[0];
        float sp = scal[2 * 64 + tid] + pb2[0];
        ampo[t]  = tanhf(sa);
        freqo[t] = MAX_FREQ_C / (1.0f + __expf(-sf));
        pho[t]   = PI_F * tanhf(sp);
    }
}

// =====================================================================
// Kernel 2: antenna_signals[b][a] = sum_s recv_amp * cos(total_phase)
// =====================================================================
__global__ void antenna_kernel(const float* __restrict__ ampv,
                               const float* __restrict__ freqv,
                               const float* __restrict__ phv,
                               const float* __restrict__ sens,
                               const float* __restrict__ aph,
                               float* __restrict__ antv)
{
    int pair = blockIdx.x * 8 + (threadIdx.x >> 5);
    int lane = threadIdx.x & 31;
    int b = pair >> 9;
    int a = pair & 511;
    float apos = (float)a * (4095.0f / 511.0f);
    float se = sens[a];
    float ap = aph[a];
    const float* fr = freqv + (size_t)b * Ss;
    const float* ph = phv  + (size_t)b * Ss;
    const float* am = ampv + (size_t)b * Ss;

    float acc = 0.0f;
    for (int s = lane; s < Ss; s += 32) {
        float d  = fabsf((float)s - apos);
        float tp = ph[s] + fr[s] * d * SPEED_INV + ap;
        acc += am[s] * se / (1.0f + DECAY_C * d) * __cosf(tp);
    }
#pragma unroll
    for (int m = 1; m < 32; m <<= 1) acc += __shfl_xor(acc, m, 32);
    if (lane == 0) antv[pair] = acc;
}

// =====================================================================
// Kernel 3: interference[b][s] = sum_a ant_sig * rev_amp * cos(rev_phase)
// =====================================================================
__global__ void interference_kernel(const float* __restrict__ freqv,
                                    const float* __restrict__ antv,
                                    const float* __restrict__ sens,
                                    const float* __restrict__ aph,
                                    float* __restrict__ intf)
{
    int t    = blockIdx.x * 8 + (threadIdx.x >> 5);
    int lane = threadIdx.x & 31;
    int b = t >> 12;
    int s = t & 4095;
    float fr = freqv[t];
    const float* ant = antv + (size_t)b * Aa;

    float acc = 0.0f;
#pragma unroll 4
    for (int a = lane; a < Aa; a += 32) {
        float apos = (float)a * (4095.0f / 511.0f);
        float d  = fabsf((float)s - apos);
        float rp = fr * d * SPEED_INV + aph[a];
        acc += ant[a] * sens[a] / (1.0f + DECAY_C * d) * __cosf(rp);
    }
#pragma unroll
    for (int m = 1; m < 32; m <<= 1) acc += __shfl_xor(acc, m, 32);
    if (lane == 0) intf[t] = acc;
}

// =====================================================================
// Kernel 4: y = hs + interference*proj_w + proj_b, then LayerNorm(H).
// =====================================================================
__global__ void finalize_ln_kernel(const float* __restrict__ hs,
                                   const float* __restrict__ intf,
                                   const float* __restrict__ projw,
                                   const float* __restrict__ projb,
                                   const float* __restrict__ lng,
                                   const float* __restrict__ lnb,
                                   float* __restrict__ out)
{
    int t    = blockIdx.x * 8 + (threadIdx.x >> 5);
    int lane = threadIdx.x & 31;
    const float* hrow = hs  + (size_t)t * Hh;
    float*       orow = out + (size_t)t * Hh;
    float iv = intf[t];

    float y[32];
    float s = 0.0f, s2 = 0.0f;
#pragma unroll
    for (int i = 0; i < 32; ++i) {
        int hc = lane + i * 32;
        float v = hrow[hc] + iv * projw[hc] + projb[hc];
        y[i] = v; s += v; s2 += v * v;
    }
#pragma unroll
    for (int m = 1; m < 32; m <<= 1) {
        s  += __shfl_xor(s,  m, 32);
        s2 += __shfl_xor(s2, m, 32);
    }
    float mu   = s * (1.0f / Hh);
    float var  = s2 * (1.0f / Hh) - mu * mu;
    float rstd = rsqrtf(var + EPS_LN);
#pragma unroll
    for (int i = 0; i < 32; ++i) {
        int hc = lane + i * 32;
        orow[hc] = (y[i] - mu) * rstd * lng[hc] + lnb[hc];
    }
}

// =====================================================================
extern "C" void kernel_launch(void* const* d_in, const int* in_sizes, int n_in,
                              void* d_out, int out_size, void* d_ws, size_t ws_size,
                              hipStream_t stream)
{
    const float* hs    = (const float*)d_in[0];
    const float* aw1   = (const float*)d_in[1];
    const float* ab1   = (const float*)d_in[2];
    const float* aw2   = (const float*)d_in[3];
    const float* ab2   = (const float*)d_in[4];
    const float* fw1   = (const float*)d_in[5];
    const float* fb1   = (const float*)d_in[6];
    const float* fw2   = (const float*)d_in[7];
    const float* fb2   = (const float*)d_in[8];
    const float* pw1   = (const float*)d_in[9];
    const float* pb1   = (const float*)d_in[10];
    const float* pw2   = (const float*)d_in[11];
    const float* pb2   = (const float*)d_in[12];
    const float* sens  = (const float*)d_in[13];
    const float* aph   = (const float*)d_in[14];
    const float* projw = (const float*)d_in[15];
    const float* projb = (const float*)d_in[16];
    const float* lng   = (const float*)d_in[17];
    const float* lnb   = (const float*)d_in[18];

    char* ws = (char*)d_ws;
    __bf16* hsbf = (__bf16*)ws;                          // 67,108,864 B
    __bf16* pw   = (__bf16*)(ws + (size_t)TOK * Hh * 2); // 3,145,728 B
    float* ampv  = (float*)(ws + (size_t)TOK * Hh * 2 + 3145728);
    float* freqv = ampv  + TOK;
    float* phv   = freqv + TOK;
    float* antv  = phv   + TOK;                          // B*A = 4096 floats
    float* intf  = antv  + Bb * Aa;                      // TOK floats

    cvt_hs_kernel<<<(TOK * Hh) / (256 * 8), 256, 0, stream>>>(hs, hsbf);
    pack_w1_kernel<<<384, 256, 0, stream>>>(aw1, fw1, pw1, pw);

    size_t lds = (size_t)MT * LDS_ROWB + 3 * 64 * 4; // 132,864 B of the 320 KB WGP LDS
    mlp_gemm_kernel<<<TOK / MT, 256, lds, stream>>>(
        hsbf, pw, ab1, fb1, pb1, aw2, fw2, pw2, ab2, fb2, pb2, ampv, freqv, phv);

    antenna_kernel<<<(Bb * Aa) / 8, 256, 0, stream>>>(ampv, freqv, phv, sens, aph, antv);
    interference_kernel<<<TOK / 8, 256, 0, stream>>>(freqv, antv, sens, aph, intf);
    finalize_ln_kernel<<<TOK / 8, 256, 0, stream>>>(hs, intf, projw, projb, lng, lnb,
                                                    (float*)d_out);
}
